// MultiheadDiffAttn_36009005809786
// MI455X (gfx1250) — matrix operand order repaired and checked by reading, not compiled, not run on hardware
//
#include <hip/hip_runtime.h>
#include <hip/hip_bf16.h>

typedef __attribute__((ext_vector_type(16))) _Float16 v16h;
typedef __attribute__((ext_vector_type(8)))  _Float16 v8h;
typedef __attribute__((ext_vector_type(8)))  float    v8f;
typedef __attribute__((ext_vector_type(4)))  unsigned int v4u;
typedef __attribute__((ext_vector_type(4)))  int      v4i;
typedef __attribute__((ext_vector_type(8)))  int      v8i;

#define DEVINL static __device__ __forceinline__

constexpr int S  = 1024;
constexpr int B  = 8;
constexpr int E  = 1024;
constexpr int H  = 16;
constexpr int Dh = 64;
constexpr int T  = S * B;          // 8192 tokens
constexpr float SCALE       = 0.125f;   // Dh^-0.5
constexpr float LAMBDA_INIT = 0.2f;
constexpr float EPS         = 1e-5f;

// ---------------------------------------------------------------------------
// WMMA helpers (CDNA5 wave32, V_WMMA_F32_16X16X32_F16)
// ---------------------------------------------------------------------------
DEVINL v8f wmma_f16(v16h a, v16h b, v8f c) {
  return __builtin_amdgcn_wmma_f32_16x16x32_f16(false, a, false, b,
                                                (short)0, c, false, false);
}

// A-fragment 16x32 (MxK), 16-bit: lane<16 -> row=lane, K={0..7,16..23};
// lane>=16 -> row=lane-16, K={8..15,24..31}.  Two aligned v8h loads.
DEVINL v16h a_frag(const _Float16* A, int ld) {
  const int lane = threadIdx.x & 31;
  const int m    = lane & 15;
  const int koff = (lane & 16) ? 8 : 0;
  const _Float16* r = A + (size_t)m * ld + koff;
  v8h lo = *(const v8h*)(r);
  v8h hi = *(const v8h*)(r + 16);
  v16h o;
#pragma unroll
  for (int j = 0; j < 8; ++j) { o[j] = lo[j]; o[j + 8] = hi[j]; }
  return o;
}

// B-fragment 32x16 (KxN) from row-major [N][K] source (NT GEMM):
// lane<16 -> N=lane, K={0..7,16..23}; lane>=16 -> N=lane-16, K={8..15,24..31}.
DEVINL v16h b_frag(const _Float16* Bm, int ld) {
  const int lane = threadIdx.x & 31;
  const int n    = lane & 15;
  const int koff = (lane & 16) ? 8 : 0;
  const _Float16* r = Bm + (size_t)n * ld + koff;
  v8h lo = *(const v8h*)(r);
  v8h hi = *(const v8h*)(r + 16);
  v16h o;
#pragma unroll
  for (int j = 0; j < 8; ++j) { o[j] = lo[j]; o[j + 8] = hi[j]; }
  return o;
}

DEVINL _Float16 toh(float v)    { return (_Float16)v; }
DEVINL _Float16 toh(_Float16 v) { return v; }
DEVINL void cstore(float* p, float v)    { *p = v; }
DEVINL void cstore(_Float16* p, float v) { *p = (_Float16)v; }

// ---------------------------------------------------------------------------
// TDM: DMA a [rows=64, cols=32 halves] weight tile (row stride K halves) from
// global into LDS at lds_off, padding each 16-DWORD row with 4 DWORDs so the
// LDS image matches Bs[64][40].  D# layout per CDNA5 ISA.
// ---------------------------------------------------------------------------
DEVINL void tdm_load_w_tile(unsigned int lds_off, const _Float16* gptr,
                            unsigned int K, unsigned int N) {
  unsigned long long ga = (unsigned long long)(uintptr_t)gptr;
  v4u g0;
  g0[0] = 1u;                                           // count=1 (user D#)
  g0[1] = lds_off;                                      // lds_addr [63:32]
  g0[2] = (unsigned int)(ga & 0xFFFFFFFFu);             // global_addr[31:0]
  g0[3] = (unsigned int)((ga >> 32) & 0x01FFFFFFu)      // global_addr[56:32]
        | (2u << 30);                                   // type=2 ("image")
  v8i g1;
  g1[0] = (int)((1u << 16)     // data_size = 2 bytes
              | (1u << 20)     // pad_enable
              | (3u << 22)     // pad_interval = 16 DWORDs (one 32-half row)
              | (3u << 25));   // pad_amount   = 4 DWORDs (8 halves)
  g1[1] = (int)((K & 0xFFFFu) << 16);                   // tensor_dim0[15:0]
  g1[2] = (int)(((K >> 16) & 0xFFFFu) | ((N & 0xFFFFu) << 16)); // dim0 hi, dim1 lo
  g1[3] = (int)(((N >> 16) & 0xFFFFu) | (32u << 16));   // dim1 hi, tile_dim0=32
  g1[4] = (int)64u;                                     // tile_dim1=64, tile_dim2=0
  g1[5] = (int)K;                                       // tensor_dim0_stride[31:0]
  g1[6] = 0;                                            // stride0 hi, stride1 lo
  g1[7] = 0;
  v4i z = {0, 0, 0, 0};
#if __clang_major__ >= 23
  v8i z8 = {0, 0, 0, 0, 0, 0, 0, 0};
  __builtin_amdgcn_tensor_load_to_lds(g0, g1, z, z, z8, 0);
#else
  __builtin_amdgcn_tensor_load_to_lds(g0, g1, z, z, 0);
#endif
}

// ---------------------------------------------------------------------------
// Generic NT GEMM: C[M,N] = A[M,K] * W[N,K]^T.  A fp32 or f16, C fp32 or f16.
// WG tile 128x64, K-step 32, 8 waves (4x2), wave tile 32x32 (2x2 WMMA tiles).
// Weight tile staged by the Tensor Data Mover (TENSORcnt); A tile staged by
// threads (fp32 -> f16 conversion fused into the LDS fill).
// ---------------------------------------------------------------------------
template <typename AT, typename OT>
__global__ __launch_bounds__(256) void gemm_nt(const AT* __restrict__ A,
                                               const _Float16* __restrict__ W,
                                               OT* __restrict__ C,
                                               int M, int N, int K) {
  constexpr int BM = 128, BN = 64, BK = 32, LDL = BK + 8;   // 80B rows, 16B aligned
  __shared__ _Float16 As[BM][LDL];
  __shared__ _Float16 Bs[BN][LDL];
  const int tid = threadIdx.x;
  const int m0  = blockIdx.y * BM;
  const int n0  = blockIdx.x * BN;
  const int w   = tid >> 5;
  const int wm  = (w >> 1) * 32;
  const int wn  = (w & 1) * 32;
  const int arow = tid >> 1, acol = (tid & 1) * 16;
  const unsigned int bs_lds = (unsigned int)(uintptr_t)&Bs[0][0];
  v8f acc[2][2] = {};

  for (int k0 = 0; k0 < K; k0 += BK) {
    __syncthreads();                       // previous tile fully consumed
    if (tid == 0)                          // one TDM issue per block (EXEC ignored)
      tdm_load_w_tile(bs_lds, W + (size_t)n0 * K + k0,
                      (unsigned int)K, (unsigned int)N);
    const AT* ap = A + (size_t)(m0 + arow) * K + k0 + acol;
    __builtin_prefetch(ap + BK, 0, 1);     // global_prefetch for next K tile
#pragma unroll
    for (int j = 0; j < 16; ++j) As[arow][acol + j] = toh(ap[j]);
    if (tid < 32)                          // TENSORcnt is per-wave: wave 0 waits
      __builtin_amdgcn_s_wait_tensorcnt(0);
    asm volatile("" ::: "memory");         // TDM wrote Bs behind the compiler's back
    __syncthreads();
#pragma unroll
    for (int i = 0; i < 2; ++i) {
      v16h af = a_frag(&As[wm + i * 16][0], LDL);
#pragma unroll
      for (int j = 0; j < 2; ++j) {
        v16h bf = b_frag(&Bs[wn + j * 16][0], LDL);
        acc[i][j] = wmma_f16(af, bf, acc[i][j]);
      }
    }
  }

  const int lane = tid & 31;
  const int n    = lane & 15;
  const int moff = (lane & 16) ? 8 : 0;
#pragma unroll
  for (int i = 0; i < 2; ++i)
#pragma unroll
    for (int j = 0; j < 2; ++j) {
      size_t base = (size_t)(m0 + wm + i * 16 + moff) * N + n0 + wn + j * 16 + n;
#pragma unroll
      for (int r = 0; r < 8; ++r)
        cstore(&C[base + (size_t)r * N], acc[i][j][r]);
    }
}

// ---------------------------------------------------------------------------
// Elementwise fp32 -> f16 conversion (weights)
// ---------------------------------------------------------------------------
__global__ __launch_bounds__(256) void cvt_f32_f16(const float* __restrict__ x,
                                                   _Float16* __restrict__ y, int n) {
  int i = blockIdx.x * 256 + threadIdx.x;
  if (i < n) y[i] = (_Float16)x[i];
}

// lambda_full = exp(sum(lq1*lk1)) - exp(sum(lq2*lk2)) + LAMBDA_INIT
__global__ void lambda_kernel(const float* lq1, const float* lk1,
                              const float* lq2, const float* lk2, float* out) {
  if (threadIdx.x == 0) {
    float s1 = 0.f, s2 = 0.f;
    for (int i = 0; i < Dh; ++i) { s1 += lq1[i] * lk1[i]; s2 += lq2[i] * lk2[i]; }
    out[0] = __expf(s1) - __expf(s2) + LAMBDA_INIT;
  }
}

// vT[(bh*128+c)*S + sk] = pv[(sk*B+b)*2E + h*128 + c]
__global__ __launch_bounds__(256) void transpose_v(const _Float16* __restrict__ pv,
                                                   _Float16* __restrict__ vt) {
  int i  = blockIdx.x * 256 + threadIdx.x;      // flat index into vt
  int sk = i & (S - 1);
  int c  = (i >> 10) & 127;
  int bh = i >> 17;
  int h  = bh & 15, b = bh >> 4;
  vt[i] = pv[((size_t)(sk * B + b)) * (2 * E) + h * 128 + c];
}

// ---------------------------------------------------------------------------
// Pass 1: online softmax stats (row max / row sum) for both score matrices.
// Block = 8 waves, each wave owns 16 q-rows and sweeps all Sk.
// ---------------------------------------------------------------------------
__global__ __launch_bounds__(256) void attn_stats(
    const _Float16* __restrict__ q1p, const _Float16* __restrict__ q2p,
    const _Float16* __restrict__ k1p, const _Float16* __restrict__ k2p,
    float* __restrict__ M1, float* __restrict__ L1,
    float* __restrict__ M2, float* __restrict__ L2) {
  const int bh   = blockIdx.y;
  const int b    = bh >> 4, h = bh & 15;
  const int w    = threadIdx.x >> 5;
  const int lane = threadIdx.x & 31;
  const int q0   = blockIdx.x * 128 + w * 16;
  const size_t ld    = (size_t)B * E;                       // token row stride
  const size_t qbase = (size_t)q0 * B * E + (size_t)b * E + h * Dh;

  v16h q1a[2], q2a[2];
#pragma unroll
  for (int kk = 0; kk < 2; ++kk) {
    q1a[kk] = a_frag(q1p + qbase + kk * 32, (int)ld);
    q2a[kk] = a_frag(q2p + qbase + kk * 32, (int)ld);
  }

  float m1[8], l1[8], m2[8], l2[8];
#pragma unroll
  for (int r = 0; r < 8; ++r) { m1[r] = -3.0e38f; l1[r] = 0.f; m2[r] = -3.0e38f; l2[r] = 0.f; }

  for (int kt = 0; kt < S / 16; ++kt) {
    const size_t kbase = (size_t)(kt * 16) * B * E + (size_t)b * E + h * Dh;
    v8f s1 = {}, s2 = {};
#pragma unroll
    for (int kk = 0; kk < 2; ++kk) {
      v16h b1 = b_frag(k1p + kbase + kk * 32, (int)ld);
      s1 = wmma_f16(q1a[kk], b1, s1);
      v16h b2 = b_frag(k2p + kbase + kk * 32, (int)ld);
      s2 = wmma_f16(q2a[kk], b2, s2);
    }
#pragma unroll
    for (int r = 0; r < 8; ++r) {
      float v1 = s1[r] * SCALE, v2 = s2[r] * SCALE;
      float t1 = v1, t2 = v2;
#pragma unroll
      for (int d = 1; d < 16; d <<= 1) {          // row reduce across half-wave
        t1 = fmaxf(t1, __shfl_xor(t1, d, 16));
        t2 = fmaxf(t2, __shfl_xor(t2, d, 16));
      }
      float nm1 = fmaxf(m1[r], t1), nm2 = fmaxf(m2[r], t2);
      float u1 = __expf(v1 - nm1), u2 = __expf(v2 - nm2);
#pragma unroll
      for (int d = 1; d < 16; d <<= 1) {
        u1 += __shfl_xor(u1, d, 16);
        u2 += __shfl_xor(u2, d, 16);
      }
      l1[r] = l1[r] * __expf(m1[r] - nm1) + u1;
      l2[r] = l2[r] * __expf(m2[r] - nm2) + u2;
      m1[r] = nm1; m2[r] = nm2;
    }
  }

  if ((lane & 15) == 0) {
    const int moff = (lane & 16) ? 8 : 0;
#pragma unroll
    for (int r = 0; r < 8; ++r) {
      size_t idx = (size_t)bh * S + q0 + moff + r;
      M1[idx] = m1[r]; L1[idx] = l1[r]; M2[idx] = m2[r]; L2[idx] = l2[r];
    }
  }
}

// ---------------------------------------------------------------------------
// Pass 2: recompute scores, emit att = att1 - lambda*att2 (fp32 to d_out),
// accumulate y = att @ V via WMMA (P tile staged per-wave in LDS to convert
// D-layout -> A-layout), fused RMSNorm + (1-lambda_init) scaling -> yh f16.
// ---------------------------------------------------------------------------
__global__ __launch_bounds__(256) void attn_emit(
    const _Float16* __restrict__ q1p, const _Float16* __restrict__ q2p,
    const _Float16* __restrict__ k1p, const _Float16* __restrict__ k2p,
    const _Float16* __restrict__ vt,
    const float* __restrict__ M1, const float* __restrict__ L1,
    const float* __restrict__ M2, const float* __restrict__ L2,
    const float* __restrict__ lamp, const float* __restrict__ rmsw,
    float* __restrict__ att, _Float16* __restrict__ yh) {
  __shared__ _Float16 P[8][16][40];                 // per-wave P tile, 16B-aligned rows
  const int bh   = blockIdx.y;
  const int b    = bh >> 4, h = bh & 15;
  const int w    = threadIdx.x >> 5;
  const int lane = threadIdx.x & 31;
  const int n    = lane & 15;
  const int moff = (lane & 16) ? 8 : 0;
  const int q0   = blockIdx.x * 128 + w * 16;
  const float lam = lamp[0];
  const size_t ld    = (size_t)B * E;
  const size_t qbase = (size_t)q0 * B * E + (size_t)b * E + h * Dh;

  v16h q1a[2], q2a[2];
#pragma unroll
  for (int kk = 0; kk < 2; ++kk) {
    q1a[kk] = a_frag(q1p + qbase + kk * 32, (int)ld);
    q2a[kk] = a_frag(q2p + qbase + kk * 32, (int)ld);
  }

  float m1[8], il1[8], m2[8], il2[8];
#pragma unroll
  for (int r = 0; r < 8; ++r) {
    size_t idx = (size_t)bh * S + q0 + moff + r;
    m1[r] = M1[idx]; il1[r] = 1.0f / L1[idx];
    m2[r] = M2[idx]; il2[r] = 1.0f / L2[idx];
  }

  v8f y[8] = {};
  const _Float16* vtb  = vt + (size_t)bh * 128 * S;
  float*          attb = att + ((size_t)bh * S + q0) * S;

  for (int kc = 0; kc < S / 32; ++kc) {
#pragma unroll
    for (int sub = 0; sub < 2; ++sub) {
      const int kt = kc * 2 + sub;
      const size_t kbase = (size_t)(kt * 16) * B * E + (size_t)b * E + h * Dh;
      v8f s1 = {}, s2 = {};
#pragma unroll
      for (int kk = 0; kk < 2; ++kk) {
        v16h b1 = b_frag(k1p + kbase + kk * 32, (int)ld);
        s1 = wmma_f16(q1a[kk], b1, s1);
        v16h b2 = b_frag(k2p + kbase + kk * 32, (int)ld);
        s2 = wmma_f16(q2a[kk], b2, s2);
      }
#pragma unroll
      for (int r = 0; r < 8; ++r) {
        float a = __expf(s1[r] * SCALE - m1[r]) * il1[r]
                - lam * __expf(s2[r] * SCALE - m2[r]) * il2[r];
        attb[(size_t)(moff + r) * S + kt * 16 + n] = a;   // fp32 att output
        P[w][moff + r][sub * 16 + n] = (_Float16)a;       // f16 for y WMMA
      }
    }
    // per-wave DS ops are in-order: P writes above precede these reads
    v16h af = a_frag(&P[w][0][0], 40);
#pragma unroll
    for (int nt = 0; nt < 8; ++nt) {
      v16h bv = b_frag(vtb + (size_t)(nt * 16) * S + kc * 32, S);
      y[nt] = wmma_f16(af, bv, y[nt]);
    }
  }

  // Fused RMSNorm over 2*Dh = 128 channels + (1 - lambda_init) scaling
#pragma unroll
  for (int r = 0; r < 8; ++r) {
    float ss = 0.f;
#pragma unroll
    for (int nt = 0; nt < 8; ++nt) ss += y[nt][r] * y[nt][r];
#pragma unroll
    for (int d = 1; d < 16; d <<= 1) ss += __shfl_xor(ss, d, 16);
    const float rms = rsqrtf(ss * (1.0f / 128.0f) + EPS);
    const float sc  = rms * (1.0f - LAMBDA_INIT);
    const size_t trow = ((size_t)(q0 + moff + r) * B + b) * (2 * E) + h * 128;
#pragma unroll
    for (int nt = 0; nt < 8; ++nt)
      yh[trow + nt * 16 + n] = (_Float16)(y[nt][r] * sc * rmsw[nt * 16 + n]);
  }
}

// ---------------------------------------------------------------------------
// Host-side launcher
// ---------------------------------------------------------------------------
extern "C" void kernel_launch(void* const* d_in, const int* in_sizes, int n_in,
                              void* d_out, int out_size, void* d_ws, size_t ws_size,
                              hipStream_t stream) {
  (void)in_sizes; (void)n_in; (void)out_size; (void)ws_size;
  const float* query = (const float*)d_in[0];
  const float* key   = (const float*)d_in[1];
  const float* value = (const float*)d_in[2];
  const float* Wq1   = (const float*)d_in[3];
  const float* Wq2   = (const float*)d_in[4];
  const float* Wk1   = (const float*)d_in[5];
  const float* Wk2   = (const float*)d_in[6];
  const float* Wv    = (const float*)d_in[7];
  const float* Wout  = (const float*)d_in[8];
  const float* lq1   = (const float*)d_in[9];
  const float* lk1   = (const float*)d_in[10];
  const float* lq2   = (const float*)d_in[11];
  const float* lk2   = (const float*)d_in[12];
  const float* rmsw  = (const float*)d_in[13];

  float* out = (float*)d_out;                       // [S,B,E]
  float* att = out + (size_t)S * B * E;             // [B,H,S,S]

  char* ws = (char*)d_ws;
  size_t o = 0;
  float* lam = (float*)(ws + o); o += 256;
  float* M1  = (float*)(ws + o); o += (size_t)B * H * S * 4;
  float* L1  = (float*)(ws + o); o += (size_t)B * H * S * 4;
  float* M2  = (float*)(ws + o); o += (size_t)B * H * S * 4;
  float* L2  = (float*)(ws + o); o += (size_t)B * H * S * 4;
  _Float16* wq1h  = (_Float16*)(ws + o); o += (size_t)E * E * 2;
  _Float16* wq2h  = (_Float16*)(ws + o); o += (size_t)E * E * 2;
  _Float16* wk1h  = (_Float16*)(ws + o); o += (size_t)E * E * 2;
  _Float16* wk2h  = (_Float16*)(ws + o); o += (size_t)E * E * 2;
  _Float16* wvh   = (_Float16*)(ws + o); o += (size_t)2 * E * E * 2;
  _Float16* wouth = (_Float16*)(ws + o); o += (size_t)2 * E * E * 2;
  _Float16* pq1   = (_Float16*)(ws + o); o += (size_t)T * E * 2;
  _Float16* pq2   = (_Float16*)(ws + o); o += (size_t)T * E * 2;
  _Float16* pk1   = (_Float16*)(ws + o); o += (size_t)T * E * 2;
  _Float16* pk2   = (_Float16*)(ws + o); o += (size_t)T * E * 2;
  _Float16* pv    = (_Float16*)(ws + o); o += (size_t)T * 2 * E * 2;
  _Float16* vtp   = (_Float16*)(ws + o); o += (size_t)B * H * 128 * S * 2;
  _Float16* yhp   = (_Float16*)(ws + o); o += (size_t)T * 2 * E * 2;

  // 1) weights -> f16
  cvt_f32_f16<<<(E * E) / 256, 256, 0, stream>>>(Wq1, wq1h, E * E);
  cvt_f32_f16<<<(E * E) / 256, 256, 0, stream>>>(Wq2, wq2h, E * E);
  cvt_f32_f16<<<(E * E) / 256, 256, 0, stream>>>(Wk1, wk1h, E * E);
  cvt_f32_f16<<<(E * E) / 256, 256, 0, stream>>>(Wk2, wk2h, E * E);
  cvt_f32_f16<<<(2 * E * E) / 256, 256, 0, stream>>>(Wv, wvh, 2 * E * E);
  cvt_f32_f16<<<(2 * E * E) / 256, 256, 0, stream>>>(Wout, wouth, 2 * E * E);

  // 2) lambda_full
  lambda_kernel<<<1, 32, 0, stream>>>(lq1, lk1, lq2, lk2, lam);

  // 3) projection GEMMs (NT, fp32 A -> f16 out), weight tiles via TDM
  dim3 gp(E / 64, T / 128);
  gemm_nt<float, _Float16><<<gp, 256, 0, stream>>>(query, wq1h, pq1, T, E, E);
  gemm_nt<float, _Float16><<<gp, 256, 0, stream>>>(query, wq2h, pq2, T, E, E);
  gemm_nt<float, _Float16><<<gp, 256, 0, stream>>>(key,   wk1h, pk1, T, E, E);
  gemm_nt<float, _Float16><<<gp, 256, 0, stream>>>(key,   wk2h, pk2, T, E, E);
  gemm_nt<float, _Float16><<<dim3(2 * E / 64, T / 128), 256, 0, stream>>>(
      value, wvh, pv, T, 2 * E, E);

  // 4) v transpose for contiguous B-fragments in att@V
  transpose_v<<<(B * H * 128 * S) / 256, 256, 0, stream>>>(pv, vtp);

  // 5) attention: stats pass + emit pass (att fp32 + yh f16)
  dim3 ga(S / 128, B * H);
  attn_stats<<<ga, 256, 0, stream>>>(pq1, pq2, pk1, pk2, M1, L1, M2, L2);
  attn_emit<<<ga, 256, 0, stream>>>(pq1, pq2, pk1, pk2, vtp,
                                    M1, L1, M2, L2, lam, rmsw, att, yhp);

  // 6) out = yh @ Wout^T  (f16 A -> fp32 out)
  gemm_nt<_Float16, float><<<dim3(E / 64, T / 128), 256, 0, stream>>>(
      yhp, wouth, out, T, E, 2 * E);
}
